// StrucFeaGNN_58480274702513
// MI455X (gfx1250) — matrix operand order repaired and checked by reading, not compiled, not run on hardware
//
#include <hip/hip_runtime.h>
#include <cstdint>

#define N_NODES   100000
#define N_EDGES   1600000
#define NUM_GRAPHS 512
#define IN_DIM    1024
#define IDENT     (IN_DIM - 2)   /* 1022 */
#define NTILES    (N_NODES / 16) /* 6250 */

typedef _Float16 half_t;
typedef __attribute__((ext_vector_type(16))) _Float16 v16h;
typedef __attribute__((ext_vector_type(8)))  float    v8f;

#define DEV static __device__ __forceinline__

DEV v8f wmma_f16(v16h a, v16h b, v8f c) {
  return __builtin_amdgcn_wmma_f32_16x16x32_f16(false, a, false, b, (short)0, c, false, false);
}

// 8 contiguous f32 at f0 -> halves 0..7 ; 8 contiguous f32 at f1 -> halves 8..15
DEV v16h cvt16(const float* f0, const float* f1) {
  float4 a = *(const float4*)f0;
  float4 b = *(const float4*)(f0 + 4);
  float4 c = *(const float4*)f1;
  float4 d = *(const float4*)(f1 + 4);
  v16h r;
  r[0]=(half_t)a.x;  r[1]=(half_t)a.y;  r[2]=(half_t)a.z;  r[3]=(half_t)a.w;
  r[4]=(half_t)b.x;  r[5]=(half_t)b.y;  r[6]=(half_t)b.z;  r[7]=(half_t)b.w;
  r[8]=(half_t)c.x;  r[9]=(half_t)c.y;  r[10]=(half_t)c.z; r[11]=(half_t)c.w;
  r[12]=(half_t)d.x; r[13]=(half_t)d.y; r[14]=(half_t)d.z; r[15]=(half_t)d.w;
  return r;
}

// ---------------------------------------------------------------- weight pack
// B fragment layout (dense 16x16x32 f16): lane = g*16 + n, half i -> k = 8g + i (i<8), 16 + 8g + (i-8) (i>=8)
__global__ __launch_bounds__(256) void k_pack(
    const float* __restrict__ w3, const float* __restrict__ w4,
    const float* __restrict__ w2s, const float* __restrict__ mw1,
    const float* __restrict__ mw2,
    half_t* __restrict__ w3f, half_t* __restrict__ w4f,
    half_t* __restrict__ w2sf, half_t* __restrict__ mlpf)
{
  int tid = blockIdx.x * 256 + threadIdx.x;
  if (tid < 1024) {                       // w3: 32 K-steps x 32 lanes
    int s = tid >> 5, lane = tid & 31;
    int g = lane >> 4, n = lane & 15;
    half_t* o = w3f + tid * 16;
#pragma unroll
    for (int i = 0; i < 16; ++i) {
      int k = s * 32 + (i < 8 ? g * 8 + i : 16 + g * 8 + (i - 8));
      o[i] = (k < IDENT) ? (half_t)w3[k * 16 + n] : (half_t)0.0f;
    }
  } else if (tid < 1152) {                // w4 (16x32) and pre_w2 (16x32): K=16 zero-padded to 32
    int u = tid - 1024;
    int which = u >> 6;                   // 0 = w4, 1 = w2s
    int v = u & 63;
    int t = v >> 5, lane = v & 31;
    int g = lane >> 4, n = t * 16 + (lane & 15);
    const float* src = which ? w2s : w4;
    half_t* o = (which ? w2sf : w4f) + v * 16;
#pragma unroll
    for (int i = 0; i < 16; ++i) {
      int k = (i < 8 ? g * 8 + i : 16 + g * 8 + (i - 8));
      o[i] = (k < 16) ? (half_t)src[k * 32 + n] : (half_t)0.0f;
    }
  } else if (tid < 2176) {                // mlp_w1[0],mlp_w1[1],mlp_w2[0],mlp_w2[1] (64x64)
    int u = tid - 1152;                   // 0..1023
    int mat = u >> 8;
    int rem = u & 255;                    // = t*64 + s*32 + lane
    int lane = rem & 31;
    int s = (rem >> 5) & 1;
    int t = rem >> 6;
    int g = lane >> 4, n = t * 16 + (lane & 15);
    const float* src = (mat < 2 ? mw1 : mw2) + (mat & 1) * 4096;
    half_t* o = mlpf + u * 16;
#pragma unroll
    for (int i = 0; i < 16; ++i) {
      int k = s * 32 + (i < 8 ? g * 8 + i : 16 + g * 8 + (i - 8));
      o[i] = (half_t)src[k * 64 + n];
    }
  }
}

// ---------------------------------------------------------------- node features
// Per wave: one 16-node tile. big GEMM x[:, :1022]@W3 -> relu -> @W4 -> relu (h_i, cols 0..31),
// and struc path 2->16->32 (h_s, cols 32..63). Output newx[100000][64].
__global__ __launch_bounds__(320) void k_node(
    const float* __restrict__ x,
    const half_t* __restrict__ w3f, const float* __restrict__ b3,
    const half_t* __restrict__ w4f, const float* __restrict__ b4,
    const float* __restrict__ w1s, const float* __restrict__ b1s,
    const half_t* __restrict__ w2sf, const float* __restrict__ b2s,
    float* __restrict__ newx)
{
  __shared__ float lds[10][16 * 17];
  int lane = threadIdx.x & 31;
  int wave = threadIdx.x >> 5;
  int tile = blockIdx.x * 10 + wave;     // 625 blocks * 10 waves = 6250 tiles exactly
  int g = lane >> 4, n = lane & 15;
  int row0 = tile * 16;
  float* L = lds[wave];
  const float* xrow = x + (size_t)(row0 + n) * IN_DIM;

  v8f acc = {};
  for (int s = 0; s < 31; ++s) {
    v16h a = cvt16(xrow + s * 32 + g * 8, xrow + s * 32 + 16 + g * 8);
    v16h b = *(const v16h*)(w3f + (s * 32 + lane) * 16);
    acc = wmma_f16(a, b, acc);
  }
  { // tail step: kbase = 992, valid k < 1022
    v16h a;
    const float* p = xrow + 992;
#pragma unroll
    for (int j = 0; j < 8; ++j) a[j] = (half_t)p[g * 8 + j];       // 992..1007 all valid
#pragma unroll
    for (int j = 0; j < 8; ++j) {
      int k = 16 + g * 8 + j;
      a[8 + j] = (992 + k < IDENT) ? (half_t)p[k] : (half_t)0.0f;
    }
    v16h b = *(const v16h*)(w3f + (31 * 32 + lane) * 16);
    acc = wmma_f16(a, b, acc);
  }
  // h16 = relu(acc + b3) -> LDS (D layout: lane n = col, rows r + 8*g)
  float bias3 = b3[n];
#pragma unroll
  for (int r = 0; r < 8; ++r)
    L[(r + 8 * g) * 17 + n] = fmaxf(acc[r] + bias3, 0.0f);
  __syncthreads();

  // h_i2 = relu(h16 @ W4 + b4): A from LDS, K=16 padded to 32
  v16h a2;
#pragma unroll
  for (int j = 0; j < 8; ++j) a2[j] = (half_t)L[n * 17 + g * 8 + j];
#pragma unroll
  for (int j = 0; j < 8; ++j) a2[8 + j] = (half_t)0.0f;
#pragma unroll
  for (int t = 0; t < 2; ++t) {
    v8f c = {};
    v16h b = *(const v16h*)(w4f + (t * 32 + lane) * 16);
    c = wmma_f16(a2, b, c);
    int col = t * 16 + n;
    float bias = b4[col];
#pragma unroll
    for (int r = 0; r < 8; ++r)
      newx[(size_t)(row0 + r + 8 * g) * 64 + col] = fmaxf(c[r] + bias, 0.0f);
  }
  __syncthreads();

  // struc path: t16 = relu(s @ W1 + b1) -> LDS
  {
    float s0 = xrow[IDENT + 0];
    float s1 = xrow[IDENT + 1];
#pragma unroll
    for (int j = 0; j < 8; ++j) {
      int d = g * 8 + j;
      L[n * 17 + d] = fmaxf(s0 * w1s[d] + s1 * w1s[16 + d] + b1s[d], 0.0f);
    }
  }
  __syncthreads();
  v16h a3;
#pragma unroll
  for (int j = 0; j < 8; ++j) a3[j] = (half_t)L[n * 17 + g * 8 + j];
#pragma unroll
  for (int j = 0; j < 8; ++j) a3[8 + j] = (half_t)0.0f;
#pragma unroll
  for (int t = 0; t < 2; ++t) {
    v8f c = {};
    v16h b = *(const v16h*)(w2sf + (t * 32 + lane) * 16);
    c = wmma_f16(a3, b, c);
    int col = t * 16 + n;
    float bias = b2s[col];
#pragma unroll
    for (int r = 0; r < 8; ++r)
      newx[(size_t)(row0 + r + 8 * g) * 64 + 32 + col] = fmaxf(c[r] + bias, 0.0f);
  }
}

// ---------------------------------------------------------------- zero / scatter / counts
__global__ __launch_bounds__(256) void k_zero(float* __restrict__ p, int n) {
  int i = blockIdx.x * 256 + threadIdx.x;
  if (i < n) p[i] = 0.0f;
}

__global__ __launch_bounds__(256) void k_scatter(
    const float* __restrict__ h, const int* __restrict__ src,
    const int* __restrict__ dst, float* __restrict__ agg)
{
  int tid = blockIdx.x * 256 + threadIdx.x;
  int e = tid >> 3;
  if (e >= N_EDGES) return;
  int part = tid & 7;
  int s = src[e], d = dst[e];
  const float* hp = h + (size_t)s * 64 + part * 8;
  float4 v0 = *(const float4*)hp;
  float4 v1 = *(const float4*)(hp + 4);
  float* ap = agg + (size_t)d * 64 + part * 8;
  unsafeAtomicAdd(ap + 0, v0.x); unsafeAtomicAdd(ap + 1, v0.y);
  unsafeAtomicAdd(ap + 2, v0.z); unsafeAtomicAdd(ap + 3, v0.w);
  unsafeAtomicAdd(ap + 4, v1.x); unsafeAtomicAdd(ap + 5, v1.y);
  unsafeAtomicAdd(ap + 6, v1.z); unsafeAtomicAdd(ap + 7, v1.w);
}

__global__ __launch_bounds__(256) void k_counts(const int* __restrict__ batch,
                                                int* __restrict__ cnt) {
  int i = blockIdx.x * 256 + threadIdx.x;
  if (i < N_NODES) atomicAdd(&cnt[batch[i]], 1);
}

// ---------------------------------------------------------------- GIN conv MLP
// z = h_in + agg; z@W1 + b1 -> bn -> relu -> @W2 + b2 -> outer bn -> + residual(s)
// optional write out, optional fused graph-pool via atomics.
__global__ __launch_bounds__(320) void k_mlp(
    const float* __restrict__ h_in, const float* __restrict__ agg,
    const half_t* __restrict__ w1f, const float* __restrict__ b1,
    const float* __restrict__ bn1g, const float* __restrict__ bn1b,
    const float* __restrict__ bn1m, const float* __restrict__ bn1v,
    const half_t* __restrict__ w2f, const float* __restrict__ b2,
    const float* __restrict__ bog, const float* __restrict__ bob,
    const float* __restrict__ bom, const float* __restrict__ bov,
    const float* __restrict__ res1, const float* __restrict__ res2,
    float* __restrict__ out, float* __restrict__ pooled,
    const int* __restrict__ batch)
{
  __shared__ float lds[10][16 * 65];
  int lane = threadIdx.x & 31;
  int wave = threadIdx.x >> 5;
  int tile = blockIdx.x * 10 + wave;
  int g = lane >> 4, n = lane & 15;
  int row0 = tile * 16;
  float* L = lds[wave];

  const float* hrow = h_in + (size_t)(row0 + n) * 64;
  const float* arow = agg  + (size_t)(row0 + n) * 64;
  v16h az[2];
#pragma unroll
  for (int s = 0; s < 2; ++s) {
    const float* ph0 = hrow + s * 32 + g * 8;
    const float* ph1 = hrow + s * 32 + 16 + g * 8;
    const float* pa0 = arow + s * 32 + g * 8;
    const float* pa1 = arow + s * 32 + 16 + g * 8;
    float4 h0 = *(const float4*)ph0, h1 = *(const float4*)(ph0 + 4);
    float4 h2 = *(const float4*)ph1, h3 = *(const float4*)(ph1 + 4);
    float4 q0 = *(const float4*)pa0, q1 = *(const float4*)(pa0 + 4);
    float4 q2 = *(const float4*)pa1, q3 = *(const float4*)(pa1 + 4);
    v16h r;
    r[0]=(half_t)(h0.x+q0.x);  r[1]=(half_t)(h0.y+q0.y);  r[2]=(half_t)(h0.z+q0.z);  r[3]=(half_t)(h0.w+q0.w);
    r[4]=(half_t)(h1.x+q1.x);  r[5]=(half_t)(h1.y+q1.y);  r[6]=(half_t)(h1.z+q1.z);  r[7]=(half_t)(h1.w+q1.w);
    r[8]=(half_t)(h2.x+q2.x);  r[9]=(half_t)(h2.y+q2.y);  r[10]=(half_t)(h2.z+q2.z); r[11]=(half_t)(h2.w+q2.w);
    r[12]=(half_t)(h3.x+q3.x); r[13]=(half_t)(h3.y+q3.y); r[14]=(half_t)(h3.z+q3.z); r[15]=(half_t)(h3.w+q3.w);
    az[s] = r;
  }

  // first 64x64 matmul + inner bn + relu -> LDS
#pragma unroll
  for (int t = 0; t < 4; ++t) {
    v8f c = {};
    c = wmma_f16(az[0], *(const v16h*)(w1f + ((t * 2 + 0) * 32 + lane) * 16), c);
    c = wmma_f16(az[1], *(const v16h*)(w1f + ((t * 2 + 1) * 32 + lane) * 16), c);
    int col = t * 16 + n;
    float sc = __frsqrt_rn(bn1v[col] + 1e-5f) * bn1g[col];
    float sb = bn1b[col] - bn1m[col] * sc;
    float bb = b1[col];
#pragma unroll
    for (int r = 0; r < 8; ++r)
      L[(r + 8 * g) * 65 + col] = fmaxf((c[r] + bb) * sc + sb, 0.0f);
  }
  __syncthreads();

  // second 64x64 matmul
  v16h a2f[2];
#pragma unroll
  for (int s = 0; s < 2; ++s) {
    v16h r;
#pragma unroll
    for (int j = 0; j < 8; ++j) r[j]     = (half_t)L[n * 65 + s * 32 + g * 8 + j];
#pragma unroll
    for (int j = 0; j < 8; ++j) r[8 + j] = (half_t)L[n * 65 + s * 32 + 16 + g * 8 + j];
    a2f[s] = r;
  }
  int bg[8];
#pragma unroll
  for (int r = 0; r < 8; ++r) bg[r] = pooled ? batch[row0 + r + 8 * g] : 0;

#pragma unroll
  for (int t = 0; t < 4; ++t) {
    v8f c = {};
    c = wmma_f16(a2f[0], *(const v16h*)(w2f + ((t * 2 + 0) * 32 + lane) * 16), c);
    c = wmma_f16(a2f[1], *(const v16h*)(w2f + ((t * 2 + 1) * 32 + lane) * 16), c);
    int col = t * 16 + n;
    float sc = __frsqrt_rn(bov[col] + 1e-5f) * bog[col];
    float sb = bob[col] - bom[col] * sc;
    float bb = b2[col];
#pragma unroll
    for (int r = 0; r < 8; ++r) {
      int node = row0 + r + 8 * g;
      size_t idx = (size_t)node * 64 + col;
      float val = (c[r] + bb) * sc + sb;
      val += res1[idx];
      if (res2)   val += res2[idx];
      if (out)    out[idx] = val;
      if (pooled) unsafeAtomicAdd(&pooled[(size_t)bg[r] * 64 + col], val);
    }
  }
}

// ---------------------------------------------------------------- head
__global__ __launch_bounds__(256) void k_head(
    const float* __restrict__ pooled, const int* __restrict__ cnt,
    const float* __restrict__ pw1, const float* __restrict__ pb1,
    const float* __restrict__ pw2, const float* __restrict__ pb2,
    float* __restrict__ outp)
{
  int gph = blockIdx.x * 256 + threadIdx.x;
  if (gph >= NUM_GRAPHS) return;
  float inv = 1.0f / fmaxf((float)cnt[gph], 1.0f);
  float p[64];
#pragma unroll
  for (int d = 0; d < 64; ++d) p[d] = pooled[gph * 64 + d] * inv;
  float t[16];
#pragma unroll
  for (int j = 0; j < 16; ++j) {
    float a = pb1[j];
#pragma unroll
    for (int d = 0; d < 64; ++d) a += p[d] * pw1[d * 16 + j];
    t[j] = fmaxf(a, 0.0f);
  }
  float o[7];
#pragma unroll
  for (int c = 0; c < 7; ++c) {
    float a = pb2[c];
#pragma unroll
    for (int j = 0; j < 16; ++j) a += t[j] * pw2[j * 7 + c];
    o[c] = a;
  }
  float mx = o[0];
#pragma unroll
  for (int c = 1; c < 7; ++c) mx = fmaxf(mx, o[c]);
  float sum = 0.0f;
#pragma unroll
  for (int c = 0; c < 7; ++c) sum += expf(o[c] - mx);
  float ls = logf(sum);
#pragma unroll
  for (int c = 0; c < 7; ++c) outp[gph * 7 + c] = o[c] - mx - ls;
}

// ---------------------------------------------------------------- launch
extern "C" void kernel_launch(void* const* d_in, const int* in_sizes, int n_in,
                              void* d_out, int out_size, void* d_ws, size_t ws_size,
                              hipStream_t stream)
{
  (void)in_sizes; (void)n_in; (void)out_size; (void)ws_size;
  const float* x       = (const float*)d_in[0];
  const int*   ei      = (const int*)d_in[1];
  const int*   batch   = (const int*)d_in[2];
  const float* pre_w1  = (const float*)d_in[3];
  const float* pre_b1  = (const float*)d_in[4];
  const float* pre_w2  = (const float*)d_in[5];
  const float* pre_b2  = (const float*)d_in[6];
  const float* pre_w3  = (const float*)d_in[7];
  const float* pre_b3  = (const float*)d_in[8];
  const float* pre_w4  = (const float*)d_in[9];
  const float* pre_b4  = (const float*)d_in[10];
  const float* mlp_w1  = (const float*)d_in[11];
  const float* mlp_b1  = (const float*)d_in[12];
  const float* mbn_g   = (const float*)d_in[13];
  const float* mbn_b   = (const float*)d_in[14];
  const float* mbn_m   = (const float*)d_in[15];
  const float* mbn_v   = (const float*)d_in[16];
  const float* mlp_w2  = (const float*)d_in[17];
  const float* mlp_b2  = (const float*)d_in[18];
  const float* bn_g    = (const float*)d_in[19];
  const float* bn_b    = (const float*)d_in[20];
  const float* bn_m    = (const float*)d_in[21];
  const float* bn_v    = (const float*)d_in[22];
  const float* post_w1 = (const float*)d_in[23];
  const float* post_b1 = (const float*)d_in[24];
  const float* post_w2 = (const float*)d_in[25];
  const float* post_b2 = (const float*)d_in[26];

  char* w = (char*)d_ws;                                      // all offsets 256B aligned
  float*  newx   = (float*)(w);                               // 100000*64 f32
  float*  g0     = (float*)(w + 25600000);
  float*  agg    = (float*)(w + 51200000);
  float*  pooled = (float*)(w + 76800000);                    // 512*64 f32
  int*    cnt    = (int*)  (w + 76800000 + 131072);           // 512 i32
  half_t* w3f    = (half_t*)(w + 76800000 + 131072 + 2048);   // 16384 half
  half_t* w4f    = (half_t*)((char*)w3f + 32768);             // 1024 half
  half_t* w2sf   = (half_t*)((char*)w4f + 2048);              // 1024 half
  half_t* mlpf   = (half_t*)((char*)w2sf + 2048);             // 4*4096 half

  k_pack<<<9, 256, 0, stream>>>(pre_w3, pre_w4, pre_w2, mlp_w1, mlp_w2,
                                w3f, w4f, w2sf, mlpf);
  k_node<<<625, 320, 0, stream>>>(x, w3f, pre_b3, w4f, pre_b4,
                                  pre_w1, pre_b1, w2sf, pre_b2, newx);
  k_zero<<<25000, 256, 0, stream>>>(agg, N_NODES * 64);
  k_zero<<<(32768 + 512 + 255) / 256, 256, 0, stream>>>(pooled, 32768 + 512);
  k_scatter<<<50000, 256, 0, stream>>>(newx, ei, ei + N_EDGES, agg);
  k_mlp<<<625, 320, 0, stream>>>(newx, agg,
      mlpf + 0 * 4096, mlp_b1 + 0,  mbn_g + 0,  mbn_b + 0,  mbn_m + 0,  mbn_v + 0,
      mlpf + 2 * 4096, mlp_b2 + 0,  bn_g + 0,   bn_b + 0,   bn_m + 0,   bn_v + 0,
      /*res1*/ newx, /*res2*/ nullptr, /*out*/ g0, /*pooled*/ nullptr, batch);
  k_zero<<<25000, 256, 0, stream>>>(agg, N_NODES * 64);
  k_scatter<<<50000, 256, 0, stream>>>(g0, ei, ei + N_EDGES, agg);
  k_mlp<<<625, 320, 0, stream>>>(g0, agg,
      mlpf + 1 * 4096, mlp_b1 + 64, mbn_g + 64, mbn_b + 64, mbn_m + 64, mbn_v + 64,
      mlpf + 3 * 4096, mlp_b2 + 64, bn_g + 64,  bn_b + 64,  bn_m + 64,  bn_v + 64,
      /*res1*/ g0, /*res2*/ newx, /*out*/ nullptr, /*pooled*/ pooled, batch);
  k_counts<<<(N_NODES + 255) / 256, 256, 0, stream>>>(batch, cnt);
  k_head<<<2, 256, 0, stream>>>(pooled, cnt, post_w1, post_b1, post_w2, post_b2,
                                (float*)d_out);
}
